// MegaBlocksMoeMLP_5076651344421
// MI455X (gfx1250) — matrix operand order repaired and checked by reading, not compiled, not run on hardware
//
#include <hip/hip_runtime.h>

// ---------------------------------------------------------------------------
// MoE MLP (top-2 of 8 experts), T=H=I=1024.  bf16 WMMA w/ fp32 accumulate.
// A-tiles staged to LDS via gfx1250 async-to-LDS copies (double-buffered).
// ---------------------------------------------------------------------------

#define T_TOK 1024
#define H_DIM 1024
#define I_DIM 1024
#define N_EXP 8
#define TWO_I 2048

#define KC        128                 // K-chunk staged per buffer
#define ROWSTRIDE 136                 // shorts per LDS row (128 + 8 pad -> 4-bank rotation)
#define CHUNK     (32 * ROWSTRIDE)    // shorts per buffer (32 rows)

typedef __attribute__((ext_vector_type(16))) __bf16 bf16x16;
typedef __attribute__((ext_vector_type(8)))  __bf16 bf16x8;
typedef __attribute__((ext_vector_type(8)))  float  f32x8;
typedef __attribute__((ext_vector_type(4)))  float  f32x4;

union Frag16 {
    bf16x16 v;
    bf16x8  h[2];
};

// gfx1250 async copy: 16B per lane, global -> LDS, tracked by ASYNCcnt.
#define ASYNC_B128(ldsoff, gptr)                                                   \
    asm volatile("global_load_async_to_lds_b128 %0, %1, off"                       \
                 :: "v"(ldsoff), "v"(gptr) : "memory")
#define WAIT_ASYNC0() asm volatile("s_wait_asynccnt 0x0" ::: "memory")

// Round-to-nearest-even f32 -> bf16 (bit trick; NaN edge ignored for speed).
static __device__ __forceinline__ unsigned short f2bf(float f) {
    unsigned u = __float_as_uint(f);
    unsigned r = (u + 0x7FFFu + ((u >> 16) & 1u)) >> 16;
    return (unsigned short)r;
}

// A fragment from an LDS row base (row = lane&15):
// lane<16 -> K[k+0:8)+K[k+16:24); lane>=16 -> K[k+8:16)+K[k+24:32)
static __device__ __forceinline__ Frag16 load_a_frag_lds(const unsigned short* rowp, int klocal, int lane) {
    const int koff = (lane < 16) ? 0 : 8;
    const unsigned short* p = rowp + klocal + koff;
    Frag16 f;
    f.h[0] = *reinterpret_cast<const bf16x8*>(p);
    f.h[1] = *reinterpret_cast<const bf16x8*>(p + 16);
    return f;
}

// B fragment (Kx16, col = lane&15): lane<16 -> K[0:8)+K[16:24); lane>=16 -> K[8:16)+K[24:32)
static __device__ __forceinline__ Frag16 load_b_frag(const unsigned short* colptr, int k0, int lane) {
    const int koff = (lane < 16) ? 0 : 8;
    const unsigned short* p = colptr + k0 + koff;
    Frag16 f;
    f.h[0] = *reinterpret_cast<const bf16x8*>(p);
    f.h[1] = *reinterpret_cast<const bf16x8*>(p + 16);
    return f;
}

static __device__ __forceinline__ f32x8 wmma_bf16(const Frag16& a, const Frag16& b, f32x8 c) {
    return __builtin_amdgcn_wmma_f32_16x16x32_bf16(false, a.v, false, b.v, (short)0, c, false, false);
}

// ---------------------------------------------------------------------------
// fp32 -> bf16 convert, float4 in / 2 dwords out
// ---------------------------------------------------------------------------
__global__ void cvt_f32_bf16(const float* __restrict__ in, unsigned short* __restrict__ out, int n4) {
    const int stride = gridDim.x * blockDim.x;
    for (int i = blockIdx.x * blockDim.x + threadIdx.x; i < n4; i += stride) {
        f32x4 v = reinterpret_cast<const f32x4*>(in)[i];
        uint2 o;
        o.x = (unsigned)f2bf(v.x) | ((unsigned)f2bf(v.y) << 16);
        o.y = (unsigned)f2bf(v.z) | ((unsigned)f2bf(v.w) << 16);
        reinterpret_cast<uint2*>(out)[i] = o;
    }
}

// ---------------------------------------------------------------------------
// Router: one wave32 per token. logits = x @ rw^T + rb, top-2, softmax,
// build per-expert token lists (slot order immaterial to final values).
// ---------------------------------------------------------------------------
__global__ __launch_bounds__(256)
void router_kernel(const float* __restrict__ x, const float* __restrict__ rw,
                   const float* __restrict__ rb,
                   int* __restrict__ counts, int* __restrict__ list_token,
                   int* __restrict__ list_dest, float* __restrict__ wgt) {
    const int wave = threadIdx.x >> 5;
    const int lane = threadIdx.x & 31;
    const int t = blockIdx.x * 8 + wave;

    float acc[N_EXP];
#pragma unroll
    for (int e = 0; e < N_EXP; ++e) acc[e] = 0.0f;

    const float* xp = x + (size_t)t * H_DIM;
    for (int h = lane; h < H_DIM; h += 32) {
        const float xv = xp[h];
#pragma unroll
        for (int e = 0; e < N_EXP; ++e) acc[e] += xv * rw[e * H_DIM + h];
    }
#pragma unroll
    for (int e = 0; e < N_EXP; ++e) {
#pragma unroll
        for (int off = 16; off > 0; off >>= 1) acc[e] += __shfl_down(acc[e], off, 32);
    }

    if (lane == 0) {
        float lg[N_EXP];
#pragma unroll
        for (int e = 0; e < N_EXP; ++e) lg[e] = acc[e] + rb[e];
        int i0 = 0;
#pragma unroll
        for (int e = 1; e < N_EXP; ++e) if (lg[e] > lg[i0]) i0 = e;
        int i1 = (i0 == 0) ? 1 : 0;
#pragma unroll
        for (int e = 0; e < N_EXP; ++e) if (e != i0 && lg[e] > lg[i1]) i1 = e;

        const float ex = __expf(lg[i1] - lg[i0]);   // <= 1
        const float w0 = 1.0f / (1.0f + ex);
        const float w1 = ex * w0;
        const int   idx[2] = { i0, i1 };
        const float w[2]   = { w0, w1 };
#pragma unroll
        for (int k = 0; k < 2; ++k) {
            const int e = idx[k];
            const int slot = atomicAdd(&counts[e], 1);
            list_token[e * T_TOK + slot] = t;
            list_dest[e * T_TOK + slot]  = 2 * t + k;
            wgt[2 * t + k] = w[k];
        }
    }
}

// ---------------------------------------------------------------------------
// GEMM1 + SiLU*mul. Block: 32 gathered token-rows (A staged to LDS, shared by
// all 8 waves), each wave 32 act-cols (gate+up halves). 8 WMMA / K-step.
// grid.x = E*32 (row tiles), grid.y = 4, block = 256 (8 waves)
// ---------------------------------------------------------------------------
__global__ __launch_bounds__(256)
void moe_gemm1(const unsigned short* __restrict__ x_bf,
               const unsigned short* __restrict__ w13_bf,
               const float* __restrict__ w13_bias,
               const int* __restrict__ counts,
               const int* __restrict__ list_token,
               unsigned short* __restrict__ act_bf) {
    const int e      = blockIdx.x >> 5;
    const int m_base = (blockIdx.x & 31) * 32;
    const int cnt    = counts[e];
    if (m_base >= cnt) return;

    const int wave   = threadIdx.x >> 5;
    const int lane   = threadIdx.x & 31;
    const int n_base = blockIdx.y * 256 + wave * 32;

    __shared__ unsigned short sA[2 * CHUNK];

    // --- staging setup: each thread copies 16B for rows srow and srow+16 ---
    const int srow  = threadIdx.x >> 4;       // 0..15
    const int scol  = threadIdx.x & 15;       // 16B sub-chunk within a row
    const int rA = m_base + srow;
    const int rB = rA + 16;
    const int tokA = (rA < cnt) ? list_token[e * T_TOK + rA] : 0;
    const int tokB = (rB < cnt) ? list_token[e * T_TOK + rB] : 0;
    const unsigned short* gA = x_bf + (size_t)tokA * H_DIM + scol * 8;
    const unsigned short* gB = x_bf + (size_t)tokB * H_DIM + scol * 8;
    const unsigned lbase = (unsigned)(unsigned long long)(void*)sA;
    const unsigned loffA = lbase + (unsigned)((srow * ROWSTRIDE + scol * 8) * 2);
    const unsigned loffB = lbase + (unsigned)(((srow + 16) * ROWSTRIDE + scol * 8) * 2);

    // --- B column pointers (gate + up halves) ---
    const int col0 = n_base + (lane & 15);
    const unsigned short* wb  = w13_bf + (size_t)e * TWO_I * H_DIM;
    const unsigned short* bg0 = wb + (size_t)col0 * H_DIM;
    const unsigned short* bg1 = wb + (size_t)(col0 + 16) * H_DIM;
    const unsigned short* bu0 = wb + (size_t)(col0 + I_DIM) * H_DIM;
    const unsigned short* bu1 = wb + (size_t)(col0 + 16 + I_DIM) * H_DIM;

    f32x8 ag[2][2] = {};
    f32x8 au[2][2] = {};

    // prologue: chunk 0 -> buffer 0
    ASYNC_B128(loffA, gA);
    ASYNC_B128(loffB, gB);

    for (int c = 0; c < H_DIM / KC; ++c) {
        WAIT_ASYNC0();          // own copies of chunk c done
        __syncthreads();        // everyone's copies of chunk c visible
        if (c + 1 < H_DIM / KC) {
            const unsigned bofs = (unsigned)(((c + 1) & 1) * (CHUNK * 2));
            ASYNC_B128(loffA + bofs, gA + (size_t)(c + 1) * KC);
            ASYNC_B128(loffB + bofs, gB + (size_t)(c + 1) * KC);
        }
        const unsigned short* bufp = &sA[(c & 1) * CHUNK];
        const unsigned short* pA0  = bufp + (lane & 15) * ROWSTRIDE;
        const unsigned short* pA1  = pA0 + 16 * ROWSTRIDE;
#pragma unroll
        for (int ks = 0; ks < KC / 32; ++ks) {
            const int k0 = c * KC + ks * 32;
            Frag16 A0 = load_a_frag_lds(pA0, ks * 32, lane);
            Frag16 A1 = load_a_frag_lds(pA1, ks * 32, lane);
            Frag16 B;
            B = load_b_frag(bg0, k0, lane);
            ag[0][0] = wmma_bf16(A0, B, ag[0][0]);
            ag[1][0] = wmma_bf16(A1, B, ag[1][0]);
            B = load_b_frag(bg1, k0, lane);
            ag[0][1] = wmma_bf16(A0, B, ag[0][1]);
            ag[1][1] = wmma_bf16(A1, B, ag[1][1]);
            B = load_b_frag(bu0, k0, lane);
            au[0][0] = wmma_bf16(A0, B, au[0][0]);
            au[1][0] = wmma_bf16(A1, B, au[1][0]);
            B = load_b_frag(bu1, k0, lane);
            au[0][1] = wmma_bf16(A0, B, au[0][1]);
            au[1][1] = wmma_bf16(A1, B, au[1][1]);
        }
        __syncthreads();        // all reads of this buffer done before reuse
    }

    // Epilogue: bias (one scalar per lane per N-tile), silu(gate)*up -> bf16
#pragma unroll
    for (int j = 0; j < 2; ++j) {
        const int col = col0 + 16 * j;
        const float bg = w13_bias[e * TWO_I + col];
        const float bu = w13_bias[e * TWO_I + I_DIM + col];
#pragma unroll
        for (int i = 0; i < 2; ++i) {
#pragma unroll
            for (int v = 0; v < 8; ++v) {
                const int row = m_base + 16 * i + 8 * (lane >> 4) + v;
                if (row < cnt) {
                    const float a = ag[i][j][v] + bg;
                    const float b = au[i][j][v] + bu;
                    const float s = a / (1.0f + __expf(-a));
                    act_bf[((size_t)e * T_TOK + row) * I_DIM + col] = f2bf(s * b);
                }
            }
        }
    }
}

// ---------------------------------------------------------------------------
// GEMM2: act[e] @ w2[e]^T + bias, scattered to g2buf[dest = 2t+k][H].
// Same LDS A-staging; 4 WMMA / K-step per wave.
// ---------------------------------------------------------------------------
__global__ __launch_bounds__(256)
void moe_gemm2(const unsigned short* __restrict__ act_bf,
               const unsigned short* __restrict__ w2_bf,
               const float* __restrict__ w2_bias,
               const int* __restrict__ counts,
               const int* __restrict__ list_dest,
               float* __restrict__ g2buf) {
    const int e      = blockIdx.x >> 5;
    const int m_base = (blockIdx.x & 31) * 32;
    const int cnt    = counts[e];
    if (m_base >= cnt) return;

    const int wave   = threadIdx.x >> 5;
    const int lane   = threadIdx.x & 31;
    const int n_base = blockIdx.y * 256 + wave * 32;

    __shared__ unsigned short sA[2 * CHUNK];

    const int srow = threadIdx.x >> 4;
    const int scol = threadIdx.x & 15;
    const unsigned short* gA = act_bf + ((size_t)e * T_TOK + m_base + srow) * I_DIM + scol * 8;
    const unsigned short* gB = gA + (size_t)16 * I_DIM;
    const unsigned lbase = (unsigned)(unsigned long long)(void*)sA;
    const unsigned loffA = lbase + (unsigned)((srow * ROWSTRIDE + scol * 8) * 2);
    const unsigned loffB = lbase + (unsigned)(((srow + 16) * ROWSTRIDE + scol * 8) * 2);

    const int col0 = n_base + (lane & 15);
    const unsigned short* wb = w2_bf + (size_t)e * H_DIM * I_DIM;
    const unsigned short* b0 = wb + (size_t)col0 * I_DIM;
    const unsigned short* b1 = wb + (size_t)(col0 + 16) * I_DIM;

    f32x8 acc[2][2] = {};

    ASYNC_B128(loffA, gA);
    ASYNC_B128(loffB, gB);

    for (int c = 0; c < I_DIM / KC; ++c) {
        WAIT_ASYNC0();
        __syncthreads();
        if (c + 1 < I_DIM / KC) {
            const unsigned bofs = (unsigned)(((c + 1) & 1) * (CHUNK * 2));
            ASYNC_B128(loffA + bofs, gA + (size_t)(c + 1) * KC);
            ASYNC_B128(loffB + bofs, gB + (size_t)(c + 1) * KC);
        }
        const unsigned short* bufp = &sA[(c & 1) * CHUNK];
        const unsigned short* pA0  = bufp + (lane & 15) * ROWSTRIDE;
        const unsigned short* pA1  = pA0 + 16 * ROWSTRIDE;
#pragma unroll
        for (int ks = 0; ks < KC / 32; ++ks) {
            const int k0 = c * KC + ks * 32;
            Frag16 A0 = load_a_frag_lds(pA0, ks * 32, lane);
            Frag16 A1 = load_a_frag_lds(pA1, ks * 32, lane);
            Frag16 B;
            B = load_b_frag(b0, k0, lane);
            acc[0][0] = wmma_bf16(A0, B, acc[0][0]);
            acc[1][0] = wmma_bf16(A1, B, acc[1][0]);
            B = load_b_frag(b1, k0, lane);
            acc[0][1] = wmma_bf16(A0, B, acc[0][1]);
            acc[1][1] = wmma_bf16(A1, B, acc[1][1]);
        }
        __syncthreads();
    }

#pragma unroll
    for (int j = 0; j < 2; ++j) {
        const int col  = col0 + 16 * j;
        const float bs = w2_bias[e * H_DIM + col];
#pragma unroll
        for (int i = 0; i < 2; ++i) {
#pragma unroll
            for (int v = 0; v < 8; ++v) {
                const int row = m_base + 16 * i + 8 * (lane >> 4) + v;
                if (row < cnt) {
                    const int dest = list_dest[e * T_TOK + row];
                    g2buf[(size_t)dest * H_DIM + col] = acc[i][j][v] + bs;
                }
            }
        }
    }
}

// ---------------------------------------------------------------------------
// Combine: out[t] = w0*g2[2t] + w1*g2[2t+1]
// ---------------------------------------------------------------------------
__global__ __launch_bounds__(256)
void combine_kernel(const float* __restrict__ g2buf, const float* __restrict__ wgt,
                    float* __restrict__ out) {
    const int t = blockIdx.x;
    const float w0 = wgt[2 * t];
    const float w1 = wgt[2 * t + 1];
    const int h = threadIdx.x * 4;
    const f32x4 a = *reinterpret_cast<const f32x4*>(g2buf + (size_t)(2 * t) * H_DIM + h);
    const f32x4 b = *reinterpret_cast<const f32x4*>(g2buf + (size_t)(2 * t + 1) * H_DIM + h);
    f32x4 o = a * w0 + b * w1;
    *reinterpret_cast<f32x4*>(out + (size_t)t * H_DIM + h) = o;
}

// ---------------------------------------------------------------------------
// Launch
// ---------------------------------------------------------------------------
extern "C" void kernel_launch(void* const* d_in, const int* in_sizes, int n_in,
                              void* d_out, int out_size, void* d_ws, size_t ws_size,
                              hipStream_t stream) {
    (void)in_sizes; (void)n_in; (void)out_size; (void)ws_size;
    const float* x    = (const float*)d_in[0];
    const float* rw   = (const float*)d_in[1];
    const float* rb   = (const float*)d_in[2];
    const float* w13  = (const float*)d_in[3];
    const float* w13b = (const float*)d_in[4];
    const float* w2   = (const float*)d_in[5];
    const float* w2b  = (const float*)d_in[6];
    float* out = (float*)d_out;

    // Workspace carve-up (~78 MB total)
    char* ws = (char*)d_ws;
    size_t off = 0;
    auto carve = [&](size_t bytes) -> void* {
        void* p = ws + off;
        off += (bytes + 255) & ~(size_t)255;
        return p;
    };
    unsigned short* w13_bf = (unsigned short*)carve((size_t)N_EXP * TWO_I * H_DIM * 2);
    unsigned short* w2_bf  = (unsigned short*)carve((size_t)N_EXP * H_DIM * I_DIM * 2);
    unsigned short* x_bf   = (unsigned short*)carve((size_t)T_TOK * H_DIM * 2);
    unsigned short* act_bf = (unsigned short*)carve((size_t)N_EXP * T_TOK * I_DIM * 2);
    float* g2buf           = (float*)carve((size_t)2 * T_TOK * H_DIM * 4);
    int* counts            = (int*)carve(N_EXP * sizeof(int));
    int* list_token        = (int*)carve((size_t)N_EXP * T_TOK * sizeof(int));
    int* list_dest         = (int*)carve((size_t)N_EXP * T_TOK * sizeof(int));
    float* wgt             = (float*)carve((size_t)2 * T_TOK * sizeof(float));

    // Per-call reset of expert counters (ws is not re-poisoned between replays).
    hipMemsetAsync(counts, 0, N_EXP * sizeof(int), stream);

    // fp32 -> bf16 conversions
    {
        const int n4_w13 = N_EXP * TWO_I * H_DIM / 4;
        const int n4_w2  = N_EXP * H_DIM * I_DIM / 4;
        const int n4_x   = T_TOK * H_DIM / 4;
        cvt_f32_bf16<<<4096, 256, 0, stream>>>(w13, w13_bf, n4_w13);
        cvt_f32_bf16<<<4096, 256, 0, stream>>>(w2,  w2_bf,  n4_w2);
        cvt_f32_bf16<<<1024, 256, 0, stream>>>(x,   x_bf,   n4_x);
    }

    // Router + list build
    router_kernel<<<T_TOK / 8, 256, 0, stream>>>(x, rw, rb, counts, list_token, list_dest, wgt);

    // Grouped GEMM1 (+SiLU*mul) and GEMM2
    moe_gemm1<<<dim3(N_EXP * 32, 4), 256, 0, stream>>>(x_bf, w13_bf, w13b, counts, list_token, act_bf);
    moe_gemm2<<<dim3(N_EXP * 32, 4), 256, 0, stream>>>(act_bf, w2_bf, w2b, counts, list_dest, g2buf);

    // Weighted combine
    combine_kernel<<<T_TOK, 256, 0, stream>>>(g2buf, wgt, out);
}
